// MoELayer_4784593568353
// MI455X (gfx1250) — matrix operand order repaired and checked by reading, not compiled, not run on hardware
//
#include <hip/hip_runtime.h>
#include <hip/hip_bf16.h>

// Problem constants (from reference)
#define BB 4
#define SS 2048
#define DD 1024
#define FF 2048
#define EE 8
#define KK 2
#define NTOK (BB*SS)          // 8192 tokens
#define NPAIR (NTOK*KK)       // 16384 routed (token,slot) pairs
#define MAX_TILES 160         // >= EE + NPAIR/128 = 136

typedef __attribute__((ext_vector_type(16))) __bf16 bf16x16;
typedef __attribute__((ext_vector_type(8)))  float  f32x8;

union ABFrag { bf16x16 v; uint4 q[2]; };

__device__ __forceinline__ unsigned short f2bf(float f) {
    unsigned int u = __float_as_uint(f);
    unsigned int r = u + 0x7FFFu + ((u >> 16) & 1u);   // round-to-nearest-even
    return (unsigned short)(r >> 16);
}

// ---------------------------------------------------------------- init / zero
__global__ __launch_bounds__(256) void k_zero(float* out, int* counts) {
    int i = blockIdx.x * blockDim.x + threadIdx.x;
    size_t total = (size_t)NTOK * DD;
    #pragma unroll
    for (int j = 0; j < 4; ++j) {
        size_t idx = (size_t)i * 4 + j;
        if (idx < total) out[idx] = 0.0f;
    }
    if (blockIdx.x == 0 && threadIdx.x < EE) counts[threadIdx.x] = 0;
}

// ---------------------------------------------------------------- conversions
__global__ __launch_bounds__(256) void k_conv_x(const float* x, unsigned short* xh) {
    int i = blockIdx.x * blockDim.x + threadIdx.x;
    int total = NTOK * DD;
    for (; i < total; i += gridDim.x * blockDim.x) xh[i] = f2bf(x[i]);
}

// w1t[e][f][d] = bf16( w1[e][d][f] )
__global__ __launch_bounds__(256) void k_conv_w1t(const float* w1, unsigned short* w1t) {
    int i = blockIdx.x * blockDim.x + threadIdx.x;
    int total = EE * FF * DD;
    for (; i < total; i += gridDim.x * blockDim.x) {
        int d = i & (DD - 1);
        int f = (i >> 10) & (FF - 1);
        int e = i >> 21;
        w1t[i] = f2bf(w1[((size_t)e * DD + d) * FF + f]);
    }
}

// w2t[e][d][f] = bf16( w2[e][f][d] )
__global__ __launch_bounds__(256) void k_conv_w2t(const float* w2, unsigned short* w2t) {
    int i = blockIdx.x * blockDim.x + threadIdx.x;
    int total = EE * DD * FF;
    for (; i < total; i += gridDim.x * blockDim.x) {
        int f = i & (FF - 1);
        int d = (i >> 11) & (DD - 1);
        int e = i >> 21;
        w2t[i] = f2bf(w2[((size_t)e * FF + f) * DD + d]);
    }
}

// ---------------------------------------------------------------- gating + top-2
__global__ __launch_bounds__(256) void k_gate_top2(const float* x, const float* gw,
                                                   float* val, int* sel) {
    int tok  = blockIdx.x * 8 + (threadIdx.x >> 5);
    int lane = threadIdx.x & 31;
    float acc[EE];
    #pragma unroll
    for (int e = 0; e < EE; ++e) acc[e] = 0.0f;
    const float* xr = x + (size_t)tok * DD;
    for (int d = lane; d < DD; d += 32) {
        float xv = xr[d];
        #pragma unroll
        for (int e = 0; e < EE; ++e) acc[e] += xv * gw[d * EE + e];
    }
    #pragma unroll
    for (int e = 0; e < EE; ++e) {
        float v = acc[e];
        #pragma unroll
        for (int off = 16; off > 0; off >>= 1) v += __shfl_xor(v, off, 32);
        acc[e] = v;
    }
    if (lane == 0) {
        int   s0 = 0;    float v0 = acc[0];
        #pragma unroll
        for (int e = 1; e < EE; ++e) if (acc[e] > v0) { v0 = acc[e]; s0 = e; }
        int   s1 = -1;   float v1 = -3.0e38f;
        #pragma unroll
        for (int e = 0; e < EE; ++e) if (e != s0 && acc[e] > v1) { v1 = acc[e]; s1 = e; }
        val[tok * 2 + 0] = v0;  sel[tok * 2 + 0] = s0;
        val[tok * 2 + 1] = v1;  sel[tok * 2 + 1] = s1;
    }
}

// Faithful softmax over the TOKEN axis (axis=1) per (batch, slot)
__global__ __launch_bounds__(256) void k_softmax_tok(const float* val, float* normw) {
    __shared__ float red[256];
    int b = blockIdx.x >> 1, slot = blockIdx.x & 1;
    int tid = threadIdx.x;
    int base = b * SS;
    float m = -3.0e38f;
    for (int s = tid; s < SS; s += 256) m = fmaxf(m, val[(base + s) * 2 + slot]);
    red[tid] = m; __syncthreads();
    for (int o = 128; o > 0; o >>= 1) { if (tid < o) red[tid] = fmaxf(red[tid], red[tid + o]); __syncthreads(); }
    m = red[0]; __syncthreads();
    float sum = 0.0f;
    for (int s = tid; s < SS; s += 256) sum += __expf(val[(base + s) * 2 + slot] - m);
    red[tid] = sum; __syncthreads();
    for (int o = 128; o > 0; o >>= 1) { if (tid < o) red[tid] += red[tid + o]; __syncthreads(); }
    sum = red[0]; __syncthreads();
    float inv = 1.0f / sum;
    for (int s = tid; s < SS; s += 256)
        normw[(base + s) * 2 + slot] = __expf(val[(base + s) * 2 + slot] - m) * inv;
}

// ---------------------------------------------------------------- routing
__global__ __launch_bounds__(256) void k_count(const int* sel, int* counts) {
    int p = blockIdx.x * blockDim.x + threadIdx.x;       // exactly NPAIR threads
    atomicAdd(&counts[sel[p]], 1);
}

__global__ void k_scan_tiles(const int* counts, int* cursor, int4* tiles, int* numTiles) {
    int tot = 0, nt = 0;
    for (int e = 0; e < EE; ++e) {
        cursor[e] = tot;
        int c = counts[e];
        for (int r = 0; r < c; r += 128) {
            int rows = c - r; if (rows > 128) rows = 128;
            tiles[nt] = make_int4(e, tot + r, rows, 0);
            ++nt;
        }
        tot += c;
    }
    *numTiles = nt;
}

__global__ __launch_bounds__(256) void k_scatter(const int* sel, const float* normw,
                                                 int* cursor, int* pairTok, float* pairW) {
    int p = blockIdx.x * blockDim.x + threadIdx.x;       // p = token*2 + slot
    int e = sel[p];
    int g = atomicAdd(&cursor[e], 1);
    pairTok[g] = p >> 1;
    pairW[g]   = normw[p];
}

// ---------------------------------------------------------------- FFN1: H = relu(Xg @ W1[e] + b1[e])
// A: gathered x rows (bf16), B: w1t[e][f][d] (cols contiguous). M-tile 128, N-tile 128.
__global__ __launch_bounds__(256) void k_ffn1(const unsigned short* __restrict__ xh,
                                              const unsigned short* __restrict__ w1t,
                                              const float* __restrict__ b1,
                                              const int* __restrict__ pairTok,
                                              const int4* __restrict__ tiles,
                                              const int* __restrict__ numTiles,
                                              unsigned short* __restrict__ Hh) {
    int tileId = blockIdx.y;
    if (tileId >= *numTiles) return;
    int4 t = tiles[tileId];
    int e = t.x, gBase = t.y, rows = t.z;
    int nBase = blockIdx.x * 128;

    int wave = threadIdx.x >> 5, lane = threadIdx.x & 31;
    int wm = wave & 3, wn = wave >> 2;          // 4 (M) x 2 (N) waves
    int mBase = wm * 32;
    int nWave = nBase + wn * 64;
    int lm = lane & 15, lh = lane >> 4;

    // Gathered A row pointers (2 M-fragments per wave)
    int rA0 = mBase + lm;          if (rA0 > rows - 1) rA0 = rows - 1;
    int rA1 = mBase + 16 + lm;     if (rA1 > rows - 1) rA1 = rows - 1;
    const unsigned short* a0p = xh + (size_t)pairTok[gBase + rA0] * DD + lh * 8;
    const unsigned short* a1p = xh + (size_t)pairTok[gBase + rA1] * DD + lh * 8;

    // B column pointers (4 N-fragments per wave)
    const unsigned short* bp[4];
    #pragma unroll
    for (int nf = 0; nf < 4; ++nf)
        bp[nf] = w1t + ((size_t)e * FF + (nWave + nf * 16 + lm)) * DD + lh * 8;

    f32x8 acc[2][4];
    #pragma unroll
    for (int i = 0; i < 2; ++i)
        #pragma unroll
        for (int j = 0; j < 4; ++j) acc[i][j] = (f32x8){0,0,0,0,0,0,0,0};

    for (int k = 0; k < DD; k += 32) {
        ABFrag a0, a1;
        a0.q[0] = *(const uint4*)(a0p + k);
        a0.q[1] = *(const uint4*)(a0p + k + 16);
        a1.q[0] = *(const uint4*)(a1p + k);
        a1.q[1] = *(const uint4*)(a1p + k + 16);
        #pragma unroll
        for (int nf = 0; nf < 4; ++nf) {
            ABFrag b;
            b.q[0] = *(const uint4*)(bp[nf] + k);
            b.q[1] = *(const uint4*)(bp[nf] + k + 16);
            acc[0][nf] = __builtin_amdgcn_wmma_f32_16x16x32_bf16(
                false, a0.v, false, b.v, (short)0, acc[0][nf], false, false);
            acc[1][nf] = __builtin_amdgcn_wmma_f32_16x16x32_bf16(
                false, a1.v, false, b.v, (short)0, acc[1][nf], false, false);
        }
    }

    // Epilogue: +b1, ReLU, bf16 store to Hh[g][f]
    #pragma unroll
    for (int mf = 0; mf < 2; ++mf) {
        #pragma unroll
        for (int nf = 0; nf < 4; ++nf) {
            int colG = nWave + nf * 16 + lm;
            float bias = b1[e * FF + colG];
            #pragma unroll
            for (int r = 0; r < 8; ++r) {
                int row = mBase + mf * 16 + r + lh * 8;
                if (row < rows) {
                    float h = acc[mf][nf][r] + bias;
                    h = fmaxf(h, 0.0f);
                    Hh[(size_t)(gBase + row) * FF + colG] = f2bf(h);
                }
            }
        }
    }
}

// ---------------------------------------------------------------- FFN2: out += w_pair * (H @ W2[e] + b2[e])
__global__ __launch_bounds__(256) void k_ffn2(const unsigned short* __restrict__ Hh,
                                              const unsigned short* __restrict__ w2t,
                                              const float* __restrict__ b2,
                                              const int* __restrict__ pairTok,
                                              const float* __restrict__ pairW,
                                              const int4* __restrict__ tiles,
                                              const int* __restrict__ numTiles,
                                              float* __restrict__ out) {
    int tileId = blockIdx.y;
    if (tileId >= *numTiles) return;
    int4 t = tiles[tileId];
    int e = t.x, gBase = t.y, rows = t.z;
    int nBase = blockIdx.x * 128;

    int wave = threadIdx.x >> 5, lane = threadIdx.x & 31;
    int wm = wave & 3, wn = wave >> 2;
    int mBase = wm * 32;
    int nWave = nBase + wn * 64;
    int lm = lane & 15, lh = lane >> 4;

    int rA0 = mBase + lm;          if (rA0 > rows - 1) rA0 = rows - 1;
    int rA1 = mBase + 16 + lm;     if (rA1 > rows - 1) rA1 = rows - 1;
    const unsigned short* a0p = Hh + (size_t)(gBase + rA0) * FF + lh * 8;
    const unsigned short* a1p = Hh + (size_t)(gBase + rA1) * FF + lh * 8;

    const unsigned short* bp[4];
    #pragma unroll
    for (int nf = 0; nf < 4; ++nf)
        bp[nf] = w2t + ((size_t)e * DD + (nWave + nf * 16 + lm)) * FF + lh * 8;

    f32x8 acc[2][4];
    #pragma unroll
    for (int i = 0; i < 2; ++i)
        #pragma unroll
        for (int j = 0; j < 4; ++j) acc[i][j] = (f32x8){0,0,0,0,0,0,0,0};

    for (int k = 0; k < FF; k += 32) {
        ABFrag a0, a1;
        a0.q[0] = *(const uint4*)(a0p + k);
        a0.q[1] = *(const uint4*)(a0p + k + 16);
        a1.q[0] = *(const uint4*)(a1p + k);
        a1.q[1] = *(const uint4*)(a1p + k + 16);
        #pragma unroll
        for (int nf = 0; nf < 4; ++nf) {
            ABFrag b;
            b.q[0] = *(const uint4*)(bp[nf] + k);
            b.q[1] = *(const uint4*)(bp[nf] + k + 16);
            acc[0][nf] = __builtin_amdgcn_wmma_f32_16x16x32_bf16(
                false, a0.v, false, b.v, (short)0, acc[0][nf], false, false);
            acc[1][nf] = __builtin_amdgcn_wmma_f32_16x16x32_bf16(
                false, a1.v, false, b.v, (short)0, acc[1][nf], false, false);
        }
    }

    // Epilogue: (+b2) * routing weight, atomic scatter-add (exactly 2 adds/element -> deterministic)
    #pragma unroll
    for (int mf = 0; mf < 2; ++mf) {
        #pragma unroll
        for (int nf = 0; nf < 4; ++nf) {
            int colG = nWave + nf * 16 + lm;
            float bias = b2[e * DD + colG];
            #pragma unroll
            for (int r = 0; r < 8; ++r) {
                int row = mBase + mf * 16 + r + lh * 8;
                if (row < rows) {
                    int g = gBase + row;
                    float y = (acc[mf][nf][r] + bias) * pairW[g];
                    atomicAdd(&out[(size_t)pairTok[g] * DD + colG], y);
                }
            }
        }
    }
}

// ---------------------------------------------------------------- launch
extern "C" void kernel_launch(void* const* d_in, const int* in_sizes, int n_in,
                              void* d_out, int out_size, void* d_ws, size_t ws_size,
                              hipStream_t stream) {
    const float* x   = (const float*)d_in[0];
    const float* gw  = (const float*)d_in[1];
    const float* w1  = (const float*)d_in[2];
    const float* b1  = (const float*)d_in[3];
    const float* w2  = (const float*)d_in[4];
    const float* b2  = (const float*)d_in[5];
    float* out = (float*)d_out;

    char* w = (char*)d_ws;
    unsigned short* xh   = (unsigned short*)w; w += (size_t)NTOK * DD * 2;      // 16.8 MB
    unsigned short* w1t  = (unsigned short*)w; w += (size_t)EE * FF * DD * 2;   // 33.6 MB
    unsigned short* w2t  = (unsigned short*)w; w += (size_t)EE * DD * FF * 2;   // 33.6 MB
    unsigned short* Hh   = (unsigned short*)w; w += (size_t)NPAIR * FF * 2;     // 67.1 MB
    float* val           = (float*)w;          w += (size_t)NTOK * 2 * 4;
    int*   sel           = (int*)w;            w += (size_t)NTOK * 2 * 4;
    float* normw         = (float*)w;          w += (size_t)NTOK * 2 * 4;
    int*   pairTok       = (int*)w;            w += (size_t)NPAIR * 4;
    float* pairW         = (float*)w;          w += (size_t)NPAIR * 4;
    int*   counts        = (int*)w;            w += 256;
    int*   cursor        = (int*)w;            w += 256;
    int*   numTiles      = (int*)w;            w += 256;
    int4*  tiles         = (int4*)w;           w += MAX_TILES * sizeof(int4);

    k_zero<<<dim3(8192), dim3(256), 0, stream>>>(out, counts);
    k_conv_x  <<<dim3(4096), dim3(256), 0, stream>>>(x, xh);
    k_conv_w1t<<<dim3(8192), dim3(256), 0, stream>>>(w1, w1t);
    k_conv_w2t<<<dim3(8192), dim3(256), 0, stream>>>(w2, w2t);
    k_gate_top2<<<dim3(NTOK / 8), dim3(256), 0, stream>>>(x, gw, val, sel);
    k_softmax_tok<<<dim3(BB * KK), dim3(256), 0, stream>>>(val, normw);
    k_count<<<dim3(NPAIR / 256), dim3(256), 0, stream>>>(sel, counts);
    k_scan_tiles<<<dim3(1), dim3(1), 0, stream>>>(counts, cursor, tiles, numTiles);
    k_scatter<<<dim3(NPAIR / 256), dim3(256), 0, stream>>>(sel, normw, cursor, pairTok, pairW);
    k_ffn1<<<dim3(FF / 128, 136), dim3(256), 0, stream>>>(xh, w1t, b1, pairTok, tiles, numTiles, Hh);
    k_ffn2<<<dim3(DD / 128, 136), dim3(256), 0, stream>>>(Hh, w2t, b2, pairTok, pairW, tiles, numTiles, out);
}
